// EdgeConv_50242527428999
// MI455X (gfx1250) — matrix operand order repaired and checked by reading, not compiled, not run on hardware
//
#include <hip/hip_runtime.h>
#include <hip/hip_bf16.h>
#include <math.h>

typedef __attribute__((ext_vector_type(16))) _Float16 v16h;
typedef __attribute__((ext_vector_type(8)))  float    v8f;

#define BATCH 8
#define NPTS  1024
#define KNN   30
#define PTS   (BATCH*NPTS)   // 8192
#define EPSV  1e-5f
#define SLOPE 0.2f

// ---------------------------------------------------------------------------
// x [B,3,N] -> xt [B*N, 3]
__global__ void k_transpose(const float* __restrict__ x, float* __restrict__ xt) {
    int e = blockIdx.x * blockDim.x + threadIdx.x;      // over 8*3*1024
    if (e >= BATCH * 3 * NPTS) return;
    int b = e / (3 * NPTS);
    int rem = e % (3 * NPTS);
    int c = rem / NPTS;
    int n = rem % NPTS;
    xt[((size_t)b * NPTS + n) * 3 + c] = x[e];
}

// ---------------------------------------------------------------------------
// Pack W [KROWS x COUT] (f32, row-major) into WMMA-fragment-ordered f16,
// zero-padded to KPAD rows:
//   wp[(((t*(KPAD/32)+kk)*32 + lane)*16 + e] = W[kk*32 + 16*(lane>>4) + e,
//                                                t*16 + (lane&15)]
// so a B-fragment is 32 contiguous bytes per lane (one v16h load).
__global__ void k_pack_w(const float* __restrict__ W, _Float16* __restrict__ wp,
                         int KPAD, int KROWS, int COUT) {
    int id = blockIdx.x * blockDim.x + threadIdx.x;
    int total = KPAD * COUT;
    if (id >= total) return;
    int e    = id & 15;
    int lane = (id >> 4) & 31;
    int rest = id >> 9;                 // t*(KPAD/32) + kk
    int nkk  = KPAD >> 5;
    int kk   = rest % nkk;
    int t    = rest / nkk;
    int n = lane & 15, h = lane >> 4;
    int k = kk * 32 + 16 * h + e;
    int col = t * 16 + n;
    wp[id] = (_Float16)((k < KROWS) ? W[(size_t)k * COUT + col] : 0.f);
}

// ---------------------------------------------------------------------------
// kNN: one block (256 thr) per point. dist[j] = 2*xi.xj - ||xj||^2 (row-const
// -||xi||^2 dropped; does not change per-row top-k). Iterative argmax, ties ->
// lowest index (matches jax top_k).
__global__ void __launch_bounds__(256) k_knn(const float* __restrict__ xin, int stride,
                                             int C, int* __restrict__ idx) {
    __shared__ float dist[NPTS];
    __shared__ float xi[128];
    __shared__ float rv[256];
    __shared__ int   ri[256];
    int pt = blockIdx.x;
    int b  = pt >> 10;
    int i  = pt & (NPTS - 1);
    const float* base = xin + (size_t)b * NPTS * stride;
    int tid = threadIdx.x;
    for (int c = tid; c < C; c += 256) xi[c] = base[(size_t)i * stride + c];
    __syncthreads();
    for (int j = tid; j < NPTS; j += 256) {
        const float* xj = base + (size_t)j * stride;
        float dot = 0.f, nj = 0.f;
        for (int c = 0; c < C; ++c) { float v = xj[c]; dot += xi[c] * v; nj += v * v; }
        dist[j] = 2.f * dot - nj;
    }
    __syncthreads();
    int* out = idx + (size_t)pt * KNN;
    for (int t = 0; t < KNN; ++t) {
        float bv = -__builtin_inff(); int bi = 0x7fffffff;
        for (int j = tid; j < NPTS; j += 256) {
            float v = dist[j];
            if (v > bv || (v == bv && j < bi)) { bv = v; bi = j; }
        }
        rv[tid] = bv; ri[tid] = bi;
        __syncthreads();
        for (int s = 128; s > 0; s >>= 1) {
            if (tid < s) {
                float v = rv[tid + s]; int jx = ri[tid + s];
                if (v > rv[tid] || (v == rv[tid] && jx < ri[tid])) { rv[tid] = v; ri[tid] = jx; }
            }
            __syncthreads();
        }
        if (tid == 0) { out[t] = ri[0]; dist[ri[0]] = -__builtin_inff(); }
        __syncthreads();
    }
}

// ---------------------------------------------------------------------------
// Swizzle helper: map within-tile K index kl (0..31) to (half h, frag elem e8)
// in contiguous order k = 16*h + e8, matching k_pack_w's B packing.
__device__ __forceinline__ void kl_to_frag(int kl, int& h, int& e8) {
    h = kl >> 4; e8 = kl & 15;
}

// ---------------------------------------------------------------------------
// EdgeConv GEMM via WMMA. One block (4 waves) per point.
//   A (LDS, f16, fragment-ordered): edge features [32 x KPAD],
//     rows 0..29 = cat(x_j - x_i, x_i), rows 30,31 = 0
//   B (global f16, fragment-ordered): packed W, software-pipelined loads
//   D: pre-BN h [30 x COUT]; per-channel sum/sumsq atomics + max over k.
template <int CIN, int KPAD, int COUT>
__global__ void __launch_bounds__(128) k_edgeconv(const float* __restrict__ xin, int stride,
                                                  const int* __restrict__ idx,
                                                  const _Float16* __restrict__ Wp,
                                                  float* __restrict__ out,   // xc+off, row stride 512
                                                  float* __restrict__ stats) // sum[c], sumsq[1024+c]
{
    constexpr int NKK = KPAD / 32;
    __shared__ v16h Asw[2 * NKK * 32];      // [tile][kk][lane] fragments
    __shared__ float xi[CIN];
    __shared__ int   nb[KNN];
    _Float16* Ah = (_Float16*)Asw;

    int pt = blockIdx.x;
    int b  = pt >> 10;
    const float* base = xin + (size_t)b * NPTS * stride;
    const float* ci   = xin + (size_t)pt * stride;
    int tid = threadIdx.x;
    if (tid < KNN) nb[tid] = idx[(size_t)pt * KNN + tid];
    for (int c = tid; c < CIN; c += 128) xi[c] = ci[c];
    __syncthreads();
    // build edge features, writing straight into fragment order
    for (int e = tid; e < 32 * KPAD; e += 128) {
        int j = e / KPAD, c = e % KPAD;
        float v = 0.f;
        if (j < KNN) {
            if (c < CIN)          v = base[(size_t)nb[j] * stride + c] - xi[c];
            else if (c < 2 * CIN) v = xi[c - CIN];
        }
        int tile = j >> 4, m = j & 15;
        int kk = c >> 5, kl = c & 31;
        int h, e8; kl_to_frag(kl, h, e8);
        Ah[(((tile * NKK + kk) * 32) + (m + 16 * h)) * 16 + e8] = (_Float16)v;
    }
    __syncthreads();

    int lane = tid & 31, wave = tid >> 5;
    int nn = lane & 15, hh = lane >> 4;
    const v16h* Wv = (const v16h*)Wp;
    constexpr int NT = COUT / 16;
    for (int t = wave; t < NT; t += 4) {
        const v16h* B = Wv + (size_t)(t * NKK) * 32 + lane;
        v8f acc0 = {}; v8f acc1 = {};
        v16h bm = B[0];
#pragma unroll
        for (int kk = 0; kk < NKK; ++kk) {
            v16h bn = (kk + 1 < NKK) ? B[(kk + 1) * 32] : bm;   // pipeline next B
            v16h a0 = Asw[(0 * NKK + kk) * 32 + lane];
            v16h a1 = Asw[(1 * NKK + kk) * 32 + lane];
            acc0 = __builtin_amdgcn_wmma_f32_16x16x32_f16(false, a0, false, bm, (short)0, acc0, false, false);
            acc1 = __builtin_amdgcn_wmma_f32_16x16x32_f16(false, a1, false, bm, (short)0, acc1, false, false);
            bm = bn;
        }
        // D layout: lane -> col nn, rows m = 8*hh+v (tile0), 16+8*hh+v (tile1)
        float s = 0.f, q = 0.f, mx = -__builtin_inff();
#pragma unroll
        for (int vv = 0; vv < 8; ++vv) {
            float x0 = acc0[vv];
            s += x0; q += x0 * x0; mx = fmaxf(mx, x0);
            bool valid = !(hh == 1 && vv >= 6);   // row 16+8*hh+vv < 30
            float x1 = acc1[vv];
            if (valid) { s += x1; q += x1 * x1; mx = fmaxf(mx, x1); }
        }
        s  += __shfl_xor(s, 16, 32);
        q  += __shfl_xor(q, 16, 32);
        mx  = fmaxf(mx, __shfl_xor(mx, 16, 32));
        if (lane < 16) {
            int co = t * 16 + nn;
            atomicAdd(&stats[co], s);
            atomicAdd(&stats[1024 + co], q);
            out[(size_t)pt * 512 + co] = mx;
        }
    }
}

// ---------------------------------------------------------------------------
// BN (from sums) + LeakyReLU, in place; optional f16 mirror copy (for WMMA A).
__global__ void k_bn_lrelu(float* __restrict__ data, int nrows, int ncols, int rowstride,
                           const float* __restrict__ stats, const float* __restrict__ g,
                           const float* __restrict__ bb, float cnt,
                           _Float16* __restrict__ out16, int stride16) {
    int total = nrows * ncols;
    for (int e = blockIdx.x * blockDim.x + threadIdx.x; e < total; e += gridDim.x * blockDim.x) {
        int r = e / ncols, c = e % ncols;
        float mean = stats[c] / cnt;
        float var  = stats[1024 + c] / cnt - mean * mean;
        float sc   = g[c] * rsqrtf(var + EPSV);
        float sh   = bb[c] - mean * sc;
        float x = data[(size_t)r * rowstride + c];
        float y = sc * x + sh;
        y = (y >= 0.f) ? y : SLOPE * y;
        data[(size_t)r * rowstride + c] = y;
        if (out16) out16[(size_t)r * stride16 + c] = (_Float16)y;
    }
}

// ---------------------------------------------------------------------------
// Pointwise conv5: h[8192,1024] = xc16[8192,512] @ W5 (fragment-packed f16).
// One block per 16-point M-tile; A staged into LDS via async-to-LDS copies,
// swizzled once into fragment order, reused by 8 waves. Each wave runs two
// N-tiles per pass with software-pipelined B loads for load/WMMA overlap.
__global__ void __launch_bounds__(256) k_gemm5(const _Float16* __restrict__ xc16,
                                               const _Float16* __restrict__ Wp,
                                               float* __restrict__ hout,
                                               float* __restrict__ stats) {
    __shared__ _Float16 Araw[16 * 512];     // raw rows, 16 KB
    __shared__ v16h     Asw[16 * 32];       // [kk][lane] fragments, 16 KB
    int mt  = blockIdx.x;                   // 512 M-tiles
    int tid = threadIdx.x;

    // stage 16 rows x 512 halves (16 KB) with gfx1250 async-to-LDS copies:
    // each lane moves 16 B; ASYNCcnt-tracked, then s_wait_asynccnt 0.
    {
        const _Float16* src = xc16 + (size_t)mt * 16 * 512;
        for (int it = 0; it < 4; ++it) {
            int elt = it * 256 + tid;                         // 1024 x 16B = 16 KB
            unsigned lds_addr = (unsigned)(uintptr_t)(&Araw[elt * 8]); // low 32b of flat = LDS offset
            unsigned long long gaddr = (unsigned long long)(uintptr_t)(src + elt * 8);
            asm volatile("global_load_async_to_lds_b128 %0, %1, off"
                         :: "v"(lds_addr), "v"(gaddr) : "memory");
        }
        asm volatile("s_wait_asynccnt 0x0" ::: "memory");
    }
    __syncthreads();
    // swizzle raw rows into fragment order
    {
        _Float16* Ah = (_Float16*)Asw;
        for (int e = tid; e < 16 * 512; e += 256) {
            int m = e >> 9;                 // row 0..15
            int c = e & 511;                // k
            int kk = c >> 5, kl = c & 31;
            int h, e8; kl_to_frag(kl, h, e8);
            Ah[((kk * 32) + (m + 16 * h)) * 16 + e8] = Araw[e];
        }
    }
    __syncthreads();

    int lane = tid & 31, wave = tid >> 5;
    int nn = lane & 15, hh = lane >> 4;
    const v16h* Wv = (const v16h*)Wp;
    // 64 N-tiles, 8 waves, 2 tiles per pass -> 4 passes of (nt0, nt0+8)
#pragma unroll
    for (int i = 0; i < 4; ++i) {
        int nt0 = wave + 16 * i;
        int nt1 = nt0 + 8;
        const v16h* B0 = Wv + (size_t)(nt0 * 16) * 32 + lane;
        const v16h* B1 = Wv + (size_t)(nt1 * 16) * 32 + lane;
        // prefetch the next pass's weight tiles (clamped; branchless)
        int pf = (i < 3) ? (nt0 + 16) : nt0;
        __builtin_prefetch((const void*)(Wv + (size_t)(pf * 16) * 32 + lane), 0, 1);
        v8f acc0 = {}, acc1 = {};
        v16h b0 = B0[0], b1 = B1[0];
#pragma unroll
        for (int kk = 0; kk < 16; ++kk) {
            v16h b0n = (kk + 1 < 16) ? B0[(kk + 1) * 32] : b0;  // pipeline next B
            v16h b1n = (kk + 1 < 16) ? B1[(kk + 1) * 32] : b1;
            v16h a = Asw[kk * 32 + lane];
            acc0 = __builtin_amdgcn_wmma_f32_16x16x32_f16(false, a, false, b0, (short)0, acc0, false, false);
            acc1 = __builtin_amdgcn_wmma_f32_16x16x32_f16(false, a, false, b1, (short)0, acc1, false, false);
            b0 = b0n; b1 = b1n;
        }
        float s0 = 0.f, q0 = 0.f, s1 = 0.f, q1 = 0.f;
#pragma unroll
        for (int vv = 0; vv < 8; ++vv) {
            float x0 = acc0[vv], x1 = acc1[vv];
            s0 += x0; q0 += x0 * x0;
            s1 += x1; q1 += x1 * x1;
            size_t row = (size_t)(mt * 16 + 8 * hh + vv) * 1024;
            hout[row + nt0 * 16 + nn] = x0;
            hout[row + nt1 * 16 + nn] = x1;
        }
        s0 += __shfl_xor(s0, 16, 32);
        q0 += __shfl_xor(q0, 16, 32);
        s1 += __shfl_xor(s1, 16, 32);
        q1 += __shfl_xor(q1, 16, 32);
        if (lane < 16) {
            atomicAdd(&stats[nt0 * 16 + nn], s0);
            atomicAdd(&stats[1024 + nt0 * 16 + nn], q0);
            atomicAdd(&stats[nt1 * 16 + nn], s1);
            atomicAdd(&stats[1024 + nt1 * 16 + nn], q1);
        }
    }
}

// ---------------------------------------------------------------------------
// Global max+mean pooling over points. f[b, 0:1024]=max, f[b, 1024:2048]=mean.
__global__ void k_pool(const float* __restrict__ h, float* __restrict__ f,
                       float* __restrict__ pmax_out) {
    int b = blockIdx.x;
    for (int ch = threadIdx.x; ch < 1024; ch += blockDim.x) {
        float mx = -__builtin_inff(), s = 0.f;
        const float* p = h + (size_t)b * NPTS * 1024 + ch;
        for (int n = 0; n < NPTS; ++n) { float v = p[(size_t)n * 1024]; mx = fmaxf(mx, v); s += v; }
        f[(size_t)b * 2048 + ch]        = mx;
        f[(size_t)b * 2048 + 1024 + ch] = s * (1.f / NPTS);
        pmax_out[(size_t)b * 1024 + ch] = mx;
    }
}

// ---------------------------------------------------------------------------
// Tiny head GEMM (M=8): one thread per output element.
__global__ void k_gemm_small(const float* __restrict__ A, const float* __restrict__ W,
                             const float* __restrict__ bias, float* __restrict__ out,
                             int M, int Kd, int Nd) {
    int e = blockIdx.x * blockDim.x + threadIdx.x;
    if (e >= M * Nd) return;
    int m = e / Nd, n = e % Nd;
    float acc = bias ? bias[n] : 0.f;
    for (int k = 0; k < Kd; ++k) acc += A[(size_t)m * Kd + k] * W[(size_t)k * Nd + n];
    out[e] = acc;
}

// BN over batch axis (8 rows) + LeakyReLU, in place.
__global__ void k_bn_head(float* __restrict__ data, const float* __restrict__ g,
                          const float* __restrict__ bb, int Nd) {
    int c = blockIdx.x * blockDim.x + threadIdx.x;
    if (c >= Nd) return;
    float s = 0.f, q = 0.f;
    for (int m = 0; m < BATCH; ++m) { float v = data[(size_t)m * Nd + c]; s += v; q += v * v; }
    float mean = s / BATCH;
    float var  = q / BATCH - mean * mean;
    float sc = g[c] * rsqrtf(var + EPSV);
    float sh = bb[c] - mean * sc;
    for (int m = 0; m < BATCH; ++m) {
        float y = sc * data[(size_t)m * Nd + c] + sh;
        data[(size_t)m * Nd + c] = (y >= 0.f) ? y : SLOPE * y;
    }
}

__global__ void k_logsoftmax(const float* __restrict__ logits, float* __restrict__ out) {
    int m = threadIdx.x;
    if (m >= BATCH) return;
    const float* r = logits + m * 40;
    float mx = -__builtin_inff();
    for (int i = 0; i < 40; ++i) mx = fmaxf(mx, r[i]);
    float s = 0.f;
    for (int i = 0; i < 40; ++i) s += expf(r[i] - mx);
    float lse = mx + logf(s);
    for (int i = 0; i < 40; ++i) out[m * 40 + i] = r[i] - lse;
}

// ===========================================================================
extern "C" void kernel_launch(void* const* d_in, const int* in_sizes, int n_in,
                              void* d_out, int out_size, void* d_ws, size_t ws_size,
                              hipStream_t stream) {
    const float* x   = (const float*)d_in[0];
    const float* W1  = (const float*)d_in[2];
    const float* g1  = (const float*)d_in[3];  const float* b1 = (const float*)d_in[4];
    const float* W2  = (const float*)d_in[5];
    const float* g2  = (const float*)d_in[6];  const float* b2 = (const float*)d_in[7];
    const float* W3  = (const float*)d_in[8];
    const float* g3  = (const float*)d_in[9];  const float* b3 = (const float*)d_in[10];
    const float* W4  = (const float*)d_in[11];
    const float* g4  = (const float*)d_in[12]; const float* b4 = (const float*)d_in[13];
    const float* W5  = (const float*)d_in[14];
    const float* g5  = (const float*)d_in[15]; const float* b5 = (const float*)d_in[16];
    const float* fW1 = (const float*)d_in[17];
    const float* fg1 = (const float*)d_in[18]; const float* fb1 = (const float*)d_in[19];
    const float* fW2 = (const float*)d_in[20];
    const float* fg2 = (const float*)d_in[21]; const float* fb2 = (const float*)d_in[22];
    const float* fW3 = (const float*)d_in[23]; const float* fb3 = (const float*)d_in[24];
    float* out = (float*)d_out;   // [8*40 logsoftmax][8*1024 p_max]

    // ---- workspace carve-up (bytes, 256-aligned) ----
    char* ws = (char*)d_ws;
    size_t off = 0;
    auto carve = [&](size_t bytes) -> char* {
        char* p = ws + off;
        off = (off + bytes + 255) & ~(size_t)255;
        return p;
    };
    float*     xt     = (float*)carve((size_t)PTS * 3 * 4);
    int*       idx    = (int*)  carve((size_t)PTS * KNN * 4);
    float*     xc     = (float*)carve((size_t)PTS * 512 * 4);
    _Float16*  xc16   = (_Float16*)carve((size_t)PTS * 512 * 2);
    float*     h5     = (float*)carve((size_t)PTS * 1024 * 4);
    _Float16*  w1p    = (_Float16*)carve((size_t)32 * 64 * 2);
    _Float16*  w2p    = (_Float16*)carve((size_t)128 * 64 * 2);
    _Float16*  w3p    = (_Float16*)carve((size_t)128 * 128 * 2);
    _Float16*  w4p    = (_Float16*)carve((size_t)256 * 256 * 2);
    _Float16*  w5p    = (_Float16*)carve((size_t)512 * 1024 * 2);
    float*     stats  = (float*)carve((size_t)5 * 2048 * 4);   // per layer: sum[c], sumsq[1024+c]
    float*     fbuf   = (float*)carve((size_t)BATCH * 2048 * 4);
    float*     h1     = (float*)carve((size_t)BATCH * 512 * 4);
    float*     h2     = (float*)carve((size_t)BATCH * 256 * 4);
    float*     logits = (float*)carve((size_t)BATCH * 40 * 4);
    (void)ws_size; (void)n_in; (void)in_sizes; (void)out_size;

    // ---- prep ----
    k_transpose<<<(BATCH * 3 * NPTS + 255) / 256, 256, 0, stream>>>(x, xt);
    k_pack_w<<<(32 * 64 + 255) / 256, 256, 0, stream>>>(W1, w1p, 32, 6, 64);
    k_pack_w<<<(128 * 64 + 255) / 256, 256, 0, stream>>>(W2, w2p, 128, 128, 64);
    k_pack_w<<<(128 * 128 + 255) / 256, 256, 0, stream>>>(W3, w3p, 128, 128, 128);
    k_pack_w<<<(256 * 256 + 255) / 256, 256, 0, stream>>>(W4, w4p, 256, 256, 256);
    k_pack_w<<<(512 * 1024 + 255) / 256, 256, 0, stream>>>(W5, w5p, 512, 512, 1024);
    hipMemsetAsync(stats, 0, (size_t)5 * 2048 * 4, stream);

    const float edge_cnt = (float)PTS * KNN;   // BN count over (B, N, k)

    // ---- layer 1: 3 -> 64, input xt (stride 3), output xc[:, 0:64] ----
    k_knn<<<PTS, 256, 0, stream>>>(xt, 3, 3, idx);
    k_edgeconv<3, 32, 64><<<PTS, 128, 0, stream>>>(xt, 3, idx, w1p, xc + 0, stats + 0 * 2048);
    k_bn_lrelu<<<1024, 256, 0, stream>>>(xc + 0, PTS, 64, 512, stats + 0 * 2048, g1, b1, edge_cnt,
                                         xc16 + 0, 512);

    // ---- layer 2: 64 -> 64, input xc[:,0:64], output xc[:,64:128] ----
    k_knn<<<PTS, 256, 0, stream>>>(xc + 0, 512, 64, idx);
    k_edgeconv<64, 128, 64><<<PTS, 128, 0, stream>>>(xc + 0, 512, idx, w2p, xc + 64, stats + 1 * 2048);
    k_bn_lrelu<<<1024, 256, 0, stream>>>(xc + 64, PTS, 64, 512, stats + 1 * 2048, g2, b2, edge_cnt,
                                         xc16 + 64, 512);

    // ---- layer 3: 64 -> 128, input xc[:,64:128], output xc[:,128:256] ----
    k_knn<<<PTS, 256, 0, stream>>>(xc + 64, 512, 64, idx);
    k_edgeconv<64, 128, 128><<<PTS, 128, 0, stream>>>(xc + 64, 512, idx, w3p, xc + 128, stats + 2 * 2048);
    k_bn_lrelu<<<2048, 256, 0, stream>>>(xc + 128, PTS, 128, 512, stats + 2 * 2048, g3, b3, edge_cnt,
                                         xc16 + 128, 512);

    // ---- layer 4: 128 -> 256, input xc[:,128:256], output xc[:,256:512] ----
    k_knn<<<PTS, 256, 0, stream>>>(xc + 128, 512, 128, idx);
    k_edgeconv<128, 256, 256><<<PTS, 128, 0, stream>>>(xc + 128, 512, idx, w4p, xc + 256, stats + 3 * 2048);
    k_bn_lrelu<<<4096, 256, 0, stream>>>(xc + 256, PTS, 256, 512, stats + 3 * 2048, g4, b4, edge_cnt,
                                         xc16 + 256, 512);

    // ---- conv5: [8192,512] @ [512,1024] via WMMA + async-to-LDS A staging ----
    k_gemm5<<<PTS / 16, 256, 0, stream>>>(xc16, w5p, h5, stats + 4 * 2048);
    k_bn_lrelu<<<8192, 256, 0, stream>>>(h5, PTS, 1024, 1024, stats + 4 * 2048, g5, b5, (float)PTS,
                                         nullptr, 0);

    // ---- pooling (also emits p_max to d_out[320:]) ----
    k_pool<<<BATCH, 256, 0, stream>>>(h5, fbuf, out + BATCH * 40);

    // ---- MLP head ----
    k_gemm_small<<<(BATCH * 512 + 255) / 256, 256, 0, stream>>>(fbuf, fW1, nullptr, h1, BATCH, 2048, 512);
    k_bn_head<<<(512 + 255) / 256, 256, 0, stream>>>(h1, fg1, fb1, 512);
    k_gemm_small<<<(BATCH * 256 + 255) / 256, 256, 0, stream>>>(h1, fW2, nullptr, h2, BATCH, 512, 256);
    k_bn_head<<<(256 + 255) / 256, 256, 0, stream>>>(h2, fg2, fb2, 256);
    k_gemm_small<<<(BATCH * 40 + 255) / 256, 256, 0, stream>>>(h2, fW3, fb3, logits, BATCH, 256, 40);
    k_logsoftmax<<<1, 64, 0, stream>>>(logits, out);
}
